// MSIGN_52518860095503
// MI455X (gfx1250) — matrix operand order repaired and checked by reading, not compiled, not run on hardware
//
#include <hip/hip_runtime.h>

// ---------------------------------------------------------------------------
// SIGN-style bidirectional graph attention, MI455X (gfx1250, wave32, WMMA).
// Memory-bound workload: all projected H=64 feature maps stored as bf16 to
// halve gather traffic; GEMMs via v_wmma_f32_16x16x32_bf16 (f32 accumulate).
// f32->bf16 via native __bf16 casts so the backend emits v_cvt_pk_bf16_f32
// instead of a 3-op manual RNE sequence.
// ---------------------------------------------------------------------------

#define N_NODE 100000
#define N_EDGE 500000
#define N_GRAPH 256
#define DN 64
#define DE 32
#define HF 64
#define DFC 200

typedef __attribute__((ext_vector_type(16))) __bf16 v16bf;
typedef __attribute__((ext_vector_type(8)))  float  v8f;

union BFrag { v16bf v; __bf16 b[16]; unsigned short us[16]; uint4 q[2]; };

__device__ __forceinline__ float bf2f(unsigned h) {
    return __uint_as_float(h << 16);
}
// order-preserving float<->uint mapping for atomic segment-max
__device__ __forceinline__ unsigned encf(float x) {
    unsigned u = __float_as_uint(x);
    return (u >> 31) ? ~u : (u | 0x80000000u);
}
__device__ __forceinline__ float decf(unsigned u) {
    return __uint_as_float((u >> 31) ? (u & 0x7FFFFFFFu) : ~u);
}

// ---------------------------------------------------------------------------
// Kernel: zero fill
// ---------------------------------------------------------------------------
__global__ void fill_zero_kernel(unsigned* __restrict__ p, long n) {
    long i = (long)blockIdx.x * blockDim.x + threadIdx.x;
    long stride = (long)gridDim.x * blockDim.x;
    for (; i < n; i += stride) p[i] = 0u;
}

// ---------------------------------------------------------------------------
// Kernel A: node projection, 4 fused weight matrices.
//   X[N,64] f32  ->  Ok[N,64] bf16 = bf16(X @ Wk + bk),  k = 0..3
// Block: 256 threads = 8 waves, 32 rows/block. Wave w: weight w>>1, mtile w&1.
// ---------------------------------------------------------------------------
__global__ void __launch_bounds__(256) node_proj_kernel(
    const float* __restrict__ X,
    const float* __restrict__ W0, const float* __restrict__ B0,
    const float* __restrict__ W1, const float* __restrict__ B1,
    const float* __restrict__ W2, const float* __restrict__ B2,
    const float* __restrict__ W3, const float* __restrict__ B3,
    __bf16* __restrict__ O0, __bf16* __restrict__ O1,
    __bf16* __restrict__ O2, __bf16* __restrict__ O3)
{
    __shared__ __align__(16) __bf16 sW[4][64][64];  // W^T, bf16: [wid][n][k]
    __shared__ float sB[4][64];

    const int tid = threadIdx.x;
    const float* Ws[4] = {W0, W1, W2, W3};
    const float* Bs[4] = {B0, B1, B2, B3};

    for (int idx = tid; idx < 4 * 64 * 64; idx += 256) {
        int wid = idx >> 12, rem = idx & 4095;
        int n = rem >> 6, k = rem & 63;
        sW[wid][n][k] = (__bf16)(Ws[wid][k * 64 + n]);
    }
    if (tid < 256) {
        int wid = tid >> 6, n = tid & 63;
        sB[wid][n] = Bs[wid][n];
    }
    __syncthreads();

    const int wave = tid >> 5, lane = tid & 31;
    const int wid = wave >> 1, mtile = wave & 1;
    const int m_block = blockIdx.x * 32;
    const int rowA = m_block + mtile * 16 + (lane & 15);
    const int kb = (lane >> 4) * 8;   // A-frag K sub-base per documented layout

    v8f acc[4];
#pragma unroll
    for (int n = 0; n < 4; n++)
#pragma unroll
        for (int r = 0; r < 8; r++) acc[n][r] = 0.0f;

#pragma unroll
    for (int kstep = 0; kstep < 2; kstep++) {
        const float* ap = X + (size_t)rowA * 64 + kstep * 32 + kb;
        float4 a0 = *(const float4*)(ap);
        float4 a1 = *(const float4*)(ap + 4);
        float4 a2 = *(const float4*)(ap + 16);
        float4 a3 = *(const float4*)(ap + 20);
        BFrag aF;
        aF.b[0]  = (__bf16)a0.x; aF.b[1]  = (__bf16)a0.y;
        aF.b[2]  = (__bf16)a0.z; aF.b[3]  = (__bf16)a0.w;
        aF.b[4]  = (__bf16)a1.x; aF.b[5]  = (__bf16)a1.y;
        aF.b[6]  = (__bf16)a1.z; aF.b[7]  = (__bf16)a1.w;
        aF.b[8]  = (__bf16)a2.x; aF.b[9]  = (__bf16)a2.y;
        aF.b[10] = (__bf16)a2.z; aF.b[11] = (__bf16)a2.w;
        aF.b[12] = (__bf16)a3.x; aF.b[13] = (__bf16)a3.y;
        aF.b[14] = (__bf16)a3.z; aF.b[15] = (__bf16)a3.w;

#pragma unroll
        for (int nt = 0; nt < 4; nt++) {
            BFrag bF;
            int n = nt * 16 + (lane & 15);
            int kb2 = kstep * 32 + (lane >> 4) * 8;
            bF.q[0] = *(const uint4*)&sW[wid][n][kb2];
            bF.q[1] = *(const uint4*)&sW[wid][n][kb2 + 16];
            acc[nt] = __builtin_amdgcn_wmma_f32_16x16x32_bf16(
                false, aF.v, false, bF.v, (short)0, acc[nt], false, false);
        }
    }

    __bf16* Os[4] = {O0, O1, O2, O3};
    __bf16* Op = Os[wid];
    const int mbase = m_block + mtile * 16 + (lane >> 4) * 8;
#pragma unroll
    for (int nt = 0; nt < 4; nt++) {
        int ncol = nt * 16 + (lane & 15);
        float bias = sB[wid][ncol];
#pragma unroll
        for (int r = 0; r < 8; r++)
            Op[(size_t)(mbase + r) * 64 + ncol] = (__bf16)(acc[nt][r] + bias);
    }
}

// ---------------------------------------------------------------------------
// Kernel B: edge projection, 2 fused weight matrices.  X[E,32] f32.
// Wave w: weight w>>2, mtile (w>>1)&1, N-half w&1. One K=32 step.
// ---------------------------------------------------------------------------
__global__ void __launch_bounds__(256) edge_proj_kernel(
    const float* __restrict__ X,
    const float* __restrict__ W0, const float* __restrict__ B0,
    const float* __restrict__ W1, const float* __restrict__ B1,
    __bf16* __restrict__ O0, __bf16* __restrict__ O1)
{
    __shared__ __align__(16) __bf16 sW[2][64][32];  // W^T bf16
    __shared__ float sB[2][64];

    const int tid = threadIdx.x;
    const float* Ws[2] = {W0, W1};
    const float* Bs[2] = {B0, B1};

    for (int idx = tid; idx < 2 * 64 * 32; idx += 256) {
        int wid = idx >> 11, rem = idx & 2047;
        int n = rem >> 5, k = rem & 31;
        sW[wid][n][k] = (__bf16)(Ws[wid][k * 64 + n]);
    }
    if (tid < 128) {
        int wid = tid >> 6, n = tid & 63;
        sB[wid][n] = Bs[wid][n];
    }
    __syncthreads();

    const int wave = tid >> 5, lane = tid & 31;
    const int wid = wave >> 2, mtile = (wave >> 1) & 1, nh = wave & 1;
    const int m_block = blockIdx.x * 32;
    const int rowA = m_block + mtile * 16 + (lane & 15);
    const int kb = (lane >> 4) * 8;

    const float* ap = X + (size_t)rowA * 32 + kb;
    float4 a0 = *(const float4*)(ap);
    float4 a1 = *(const float4*)(ap + 4);
    float4 a2 = *(const float4*)(ap + 16);
    float4 a3 = *(const float4*)(ap + 20);
    BFrag aF;
    aF.b[0]  = (__bf16)a0.x; aF.b[1]  = (__bf16)a0.y;
    aF.b[2]  = (__bf16)a0.z; aF.b[3]  = (__bf16)a0.w;
    aF.b[4]  = (__bf16)a1.x; aF.b[5]  = (__bf16)a1.y;
    aF.b[6]  = (__bf16)a1.z; aF.b[7]  = (__bf16)a1.w;
    aF.b[8]  = (__bf16)a2.x; aF.b[9]  = (__bf16)a2.y;
    aF.b[10] = (__bf16)a2.z; aF.b[11] = (__bf16)a2.w;
    aF.b[12] = (__bf16)a3.x; aF.b[13] = (__bf16)a3.y;
    aF.b[14] = (__bf16)a3.z; aF.b[15] = (__bf16)a3.w;

    v8f acc[2];
#pragma unroll
    for (int n = 0; n < 2; n++)
#pragma unroll
        for (int r = 0; r < 8; r++) acc[n][r] = 0.0f;

#pragma unroll
    for (int t = 0; t < 2; t++) {
        int nt = nh * 2 + t;
        BFrag bF;
        int n = nt * 16 + (lane & 15);
        int kb2 = (lane >> 4) * 8;
        bF.q[0] = *(const uint4*)&sW[wid][n][kb2];
        bF.q[1] = *(const uint4*)&sW[wid][n][kb2 + 16];
        acc[t] = __builtin_amdgcn_wmma_f32_16x16x32_bf16(
            false, aF.v, false, bF.v, (short)0, acc[t], false, false);
    }

    __bf16* Os[2] = {O0, O1};
    __bf16* Op = Os[wid];
    const int mbase = m_block + mtile * 16 + (lane >> 4) * 8;
#pragma unroll
    for (int t = 0; t < 2; t++) {
        int ncol = (nh * 2 + t) * 16 + (lane & 15);
        float bias = sB[wid][ncol];
#pragma unroll
        for (int r = 0; r < 8; r++)
            Op[(size_t)(mbase + r) * 64 + ncol] = (__bf16)(acc[t][r] + bias);
    }
}

// ---------------------------------------------------------------------------
// Kernel C: per-edge attention logit + segment-max over destination node.
// ---------------------------------------------------------------------------
__device__ __forceinline__ float att2(unsigned a, unsigned b, unsigned c,
                                      float w0, float w1, float sl) {
    float x0 = bf2f(a & 0xFFFFu) + bf2f(b & 0xFFFFu) + bf2f(c & 0xFFFFu);
    float x1 = bf2f(a >> 16) + bf2f(b >> 16) + bf2f(c >> 16);
    x0 = (x0 > 0.f) ? x0 : sl * x0;
    x1 = (x1 > 0.f) ? x1 : sl * x1;
    return fmaf(x0, w0, x1 * w1);
}

__global__ void __launch_bounds__(256) attn_logit_kernel(
    const unsigned short* __restrict__ hs, const unsigned short* __restrict__ hd,
    const unsigned short* __restrict__ ep,
    const int* __restrict__ src, const int* __restrict__ dst,
    const float* __restrict__ attW, const float* __restrict__ att_a,
    const float* __restrict__ att_b,
    float* __restrict__ wlog, unsigned* __restrict__ menc, int nE)
{
    __shared__ float sW[64];
    __shared__ float sSl, sBi;
    int tid = threadIdx.x;
    if (tid < 64) sW[tid] = attW[tid];
    if (tid == 64) sSl = att_a[0];
    if (tid == 65) sBi = att_b[0];
    __syncthreads();

    int e = blockIdx.x * 256 + tid;
    if (e >= nE) return;
    int s = src[e], d = dst[e];
    const uint4* hp = (const uint4*)(hs + (size_t)s * 64);
    const uint4* dp = (const uint4*)(hd + (size_t)d * 64);
    const uint4* cp = (const uint4*)(ep + (size_t)e * 64);
    float sl = sSl, acc = 0.f;
#pragma unroll
    for (int q = 0; q < 8; q++) {
        uint4 A = hp[q], B = dp[q], C = cp[q];
        const float* w = &sW[q * 8];
        acc += att2(A.x, B.x, C.x, w[0], w[1], sl);
        acc += att2(A.y, B.y, C.y, w[2], w[3], sl);
        acc += att2(A.z, B.z, C.z, w[4], w[5], sl);
        acc += att2(A.w, B.w, C.w, w[6], w[7], sl);
    }
    float logit = acc + sBi;
    wlog[e] = logit;
    atomicMax(&menc[d], encf(logit));
}

// ---------------------------------------------------------------------------
// Kernel D: ex = exp(w - max[dst]);  segment-sum over dst.
// ---------------------------------------------------------------------------
__global__ void __launch_bounds__(256) exp_sum_kernel(
    const float* __restrict__ wlog, const int* __restrict__ dst,
    const unsigned* __restrict__ menc, float* __restrict__ exa,
    float* __restrict__ ssum, int nE)
{
    int e = blockIdx.x * 256 + threadIdx.x;
    if (e >= nE) return;
    int d = dst[e];
    float ex = __expf(wlog[e] - decf(menc[d]));
    exa[e] = ex;
    atomicAdd(&ssum[d], ex);
}

// ---------------------------------------------------------------------------
// Kernel E: value combine + graph pooling. gid is sorted, so accumulate runs
// in registers and flush one atomic per gid change. 64 threads = H lanes.
// ---------------------------------------------------------------------------
#define POOL_CHUNK 256
__global__ void __launch_bounds__(64) pool_kernel(
    const unsigned short* __restrict__ hs2, const unsigned short* __restrict__ hd2,
    const unsigned short* __restrict__ e2,
    const int* __restrict__ src, const int* __restrict__ dst,
    const int* __restrict__ gid,
    const float* __restrict__ exa, const float* __restrict__ ssum,
    float* __restrict__ pooled, int nE)
{
    int t = threadIdx.x;
    int base = blockIdx.x * POOL_CHUNK;
    if (base >= nE) return;
    int end = min(base + POOL_CHUNK, nE);
    int cur = gid[base];
    float acc = 0.f;
    for (int e = base; e < end; e++) {
        if (e + 8 < end)
            __builtin_prefetch(&e2[(size_t)(e + 8) * 64 + t], 0, 1);
        int g = gid[e];
        if (g != cur) {
            atomicAdd(&pooled[cur * 64 + t], acc);
            acc = 0.f;
            cur = g;
        }
        int s = src[e], d = dst[e];
        float a = exa[e] / ssum[d];
        float v = a * bf2f(e2[(size_t)e * 64 + t])
                    * bf2f(hs2[(size_t)s * 64 + t])
                    * bf2f(hd2[(size_t)d * 64 + t]);
        acc += v;
    }
    atomicAdd(&pooled[cur * 64 + t], acc);
}

// ---------------------------------------------------------------------------
// Kernel F: FC head (eval mode), one block per graph. Writes out[g*2 + col].
// ---------------------------------------------------------------------------
__global__ void __launch_bounds__(256) fc_head_kernel(
    const float* __restrict__ pooled,
    const float* __restrict__ fc1W, const float* __restrict__ fc1b,
    const float* __restrict__ fc2W, const float* __restrict__ fc2b,
    const float* __restrict__ fc3W, const float* __restrict__ fc3b,
    float* __restrict__ out, int col)
{
    __shared__ float sx[64];
    __shared__ float h1[DFC];
    __shared__ float h2[DFC];
    __shared__ float red[256];
    const float inv = rsqrtf(1.0f + 1e-5f);
    int tid = threadIdx.x;
    int g = blockIdx.x;
    if (tid < 64) sx[tid] = pooled[g * 64 + tid];
    __syncthreads();

    if (tid < DFC) {
        float a = fc1b[tid];
        for (int k = 0; k < 64; k++) a = fmaf(sx[k], fc1W[k * DFC + tid], a);
        a = (a > 0.f) ? a : 0.01f * a;
        h1[tid] = a * inv;
    }
    __syncthreads();
    if (tid < DFC) {
        float a = fc2b[tid];
        for (int k = 0; k < DFC; k++) a = fmaf(h1[k], fc2W[k * DFC + tid], a);
        a = (a > 0.f) ? a : 0.01f * a;
        h2[tid] = a * inv;
    }
    __syncthreads();
    red[tid] = (tid < DFC) ? h2[tid] * fc3W[tid] : 0.f;
    __syncthreads();
    for (int s = 128; s > 0; s >>= 1) {
        if (tid < s) red[tid] += red[tid + s];
        __syncthreads();
    }
    if (tid == 0) out[g * 2 + col] = red[0] + fc3b[0];
}

// ---------------------------------------------------------------------------
// Host launch
// ---------------------------------------------------------------------------
extern "C" void kernel_launch(void* const* d_in, const int* in_sizes, int n_in,
                              void* d_out, int out_size, void* d_ws, size_t ws_size,
                              hipStream_t stream) {
    const float* node_lig = (const float*)d_in[0];
    const float* node_poc = (const float*)d_in[1];
    const float* edge_f[2] = {(const float*)d_in[2], (const float*)d_in[3]};
    const int* srcI[2] = {(const int*)d_in[4], (const int*)d_in[7]};
    const int* dstI[2] = {(const int*)d_in[5], (const int*)d_in[8]};
    const int* gidI[2] = {(const int*)d_in[6], (const int*)d_in[9]};

    // params flattened as JAX pytree leaves (dict keys sorted):
    enum { L_ATT_W = 0, L_ATT_A, L_ATT_B, L_FC1_W, L_FC1_B, L_FC2_W, L_FC2_B,
           L_FC3_W, L_FC3_B, L_PRJ_DST_W, L_PRJ_DST_B, L_PRJ_EDGE_W,
           L_PRJ_EDGE_B, L_PRJ_SRC_W, L_PRJ_SRC_B, L_W_DST_W, L_W_DST_B,
           L_W_EDGE_W, L_W_EDGE_B, L_W_SRC_W, L_W_SRC_B };
    const int base[2] = {10, 31};   // params_lp, params_pl
    auto P = [&](int dir, int leaf) { return (const float*)d_in[base[dir] + leaf]; };

    // workspace bump allocator
    char* ws = (char*)d_ws;
    size_t off = 0;
    auto alloc = [&](size_t bytes) -> void* {
        void* p = ws + off;
        off = (off + bytes + 255) & ~(size_t)255;
        return p;
    };
    const size_t NB = (size_t)N_NODE * 64 * sizeof(unsigned short);
    const size_t EB = (size_t)N_EDGE * 64 * sizeof(unsigned short);

    unsigned short *hs[2], *hd[2], *hs2[2], *hd2[2], *ep[2], *ep2[2];
    float *wlog[2], *exa[2], *ssum[2], *pooled[2];
    unsigned* menc[2];
    for (int d = 0; d < 2; d++) {
        hs[d]  = (unsigned short*)alloc(NB);
        hd[d]  = (unsigned short*)alloc(NB);
        hs2[d] = (unsigned short*)alloc(NB);
        hd2[d] = (unsigned short*)alloc(NB);
        ep[d]  = (unsigned short*)alloc(EB);
        ep2[d] = (unsigned short*)alloc(EB);
        wlog[d] = (float*)alloc((size_t)N_EDGE * 4);
        exa[d]  = (float*)alloc((size_t)N_EDGE * 4);
        menc[d] = (unsigned*)alloc((size_t)N_NODE * 4);
        ssum[d] = (float*)alloc((size_t)N_NODE * 4);
        pooled[d] = (float*)alloc((size_t)N_GRAPH * 64 * 4);
    }

    for (int d = 0; d < 2; d++) {
        fill_zero_kernel<<<256, 256, 0, stream>>>(menc[d], N_NODE);
        fill_zero_kernel<<<256, 256, 0, stream>>>((unsigned*)ssum[d], N_NODE);
        fill_zero_kernel<<<64, 256, 0, stream>>>((unsigned*)pooled[d], N_GRAPH * 64);
    }

    // ---- node projections (fused: each node matrix read once, 4 GEMMs) ----
    // node_lig: lp.prj_src -> hs[0], lp.w_src -> hs2[0],
    //           pl.prj_dst -> hd[1], pl.w_dst -> hd2[1]
    node_proj_kernel<<<N_NODE / 32, 256, 0, stream>>>(
        node_lig,
        P(0, L_PRJ_SRC_W), P(0, L_PRJ_SRC_B),
        P(0, L_W_SRC_W),   P(0, L_W_SRC_B),
        P(1, L_PRJ_DST_W), P(1, L_PRJ_DST_B),
        P(1, L_W_DST_W),   P(1, L_W_DST_B),
        (__bf16*)hs[0], (__bf16*)hs2[0], (__bf16*)hd[1], (__bf16*)hd2[1]);
    // node_poc: lp.prj_dst -> hd[0], lp.w_dst -> hd2[0],
    //           pl.prj_src -> hs[1], pl.w_src -> hs2[1]
    node_proj_kernel<<<N_NODE / 32, 256, 0, stream>>>(
        node_poc,
        P(0, L_PRJ_DST_W), P(0, L_PRJ_DST_B),
        P(0, L_W_DST_W),   P(0, L_W_DST_B),
        P(1, L_PRJ_SRC_W), P(1, L_PRJ_SRC_B),
        P(1, L_W_SRC_W),   P(1, L_W_SRC_B),
        (__bf16*)hd[0], (__bf16*)hd2[0], (__bf16*)hs[1], (__bf16*)hs2[1]);

    // ---- edge projections ----
    for (int d = 0; d < 2; d++) {
        edge_proj_kernel<<<N_EDGE / 32, 256, 0, stream>>>(
            edge_f[d],
            P(d, L_PRJ_EDGE_W), P(d, L_PRJ_EDGE_B),
            P(d, L_W_EDGE_W),   P(d, L_W_EDGE_B),
            (__bf16*)ep[d], (__bf16*)ep2[d]);
    }

    // ---- attention softmax + pooling + FC head ----
    const int gE = (N_EDGE + 255) / 256;
    for (int d = 0; d < 2; d++) {
        attn_logit_kernel<<<gE, 256, 0, stream>>>(
            hs[d], hd[d], ep[d], srcI[d], dstI[d],
            P(d, L_ATT_W), P(d, L_ATT_A), P(d, L_ATT_B),
            wlog[d], menc[d], N_EDGE);
        exp_sum_kernel<<<gE, 256, 0, stream>>>(
            wlog[d], dstI[d], menc[d], exa[d], ssum[d], N_EDGE);
        pool_kernel<<<(N_EDGE + POOL_CHUNK - 1) / POOL_CHUNK, 64, 0, stream>>>(
            hs2[d], hd2[d], ep2[d], srcI[d], dstI[d], gidI[d],
            exa[d], ssum[d], pooled[d], N_EDGE);
        fc_head_kernel<<<N_GRAPH, 256, 0, stream>>>(
            pooled[d],
            P(d, L_FC1_W), P(d, L_FC1_B),
            P(d, L_FC2_W), P(d, L_FC2_B),
            P(d, L_FC3_W), P(d, L_FC3_B),
            (float*)d_out, d);
    }
}
